// DecodeYoloV1_22694607192620
// MI455X (gfx1250) — compile-verified, matrix operand order checked by reading, not run on hardware
//
#include <hip/hip_runtime.h>
#include <hip/hip_bf16.h>
#include <stdint.h>
#include <math.h>

// ---------------- constants (match reference) ----------------
#define GRID7   7
#define NCELLS  49          // 7*7
#define NCH     30          // 2*5 + 20
#define NIN     1470        // 49*30
#define NCLS    20
#define STRIDE  64.0f       // 448/7
#define CONF_TH 0.5f
#define IOU_TH  0.5f

// ---------------- vector types ----------------
typedef _Float16 v16h __attribute__((ext_vector_type(16)));
typedef float    v8f  __attribute__((ext_vector_type(8)));
typedef unsigned int u32x4 __attribute__((ext_vector_type(4)));
typedef int          i32x4 __attribute__((ext_vector_type(4)));
typedef int          i32x8 __attribute__((ext_vector_type(8)));

__global__ __launch_bounds__(256)
void yolo_decode_nms_kernel(const float* __restrict__ x, float* __restrict__ out)
{
    __shared__ float s_p[NIN];             // sigmoided activations
    __shared__ float s_box[NCELLS * 6];    // decoded boxes (cell order)
    __shared__ float s_conf[64];           // per-cell best conf (padded with -1)
    __shared__ int   s_rank[64];           // stable descending rank
    __shared__ float s_sb[NCELLS * 6];     // sorted boxes
    __shared__ float s_iou[NCELLS * NCELLS];
    __shared__ int   s_keep[NCELLS];

    const int t = threadIdx.x;

    // ---------------- Phase 1: HBM -> LDS via Tensor Data Mover ----------------
#if __has_builtin(__builtin_amdgcn_tensor_load_to_lds) && __has_builtin(__builtin_amdgcn_s_wait_tensorcnt)
    if (t == 0) {
        const uint64_t ga  = (uint64_t)(uintptr_t)x;
        const uint32_t lds = (uint32_t)(uintptr_t)&s_p[0]; // low 32 bits = LDS byte addr

        u32x4 g0;
        g0[0] = 1u;                                        // count=1, user descriptor
        g0[1] = lds;                                       // lds_addr
        g0[2] = (uint32_t)(ga & 0xFFFFFFFFu);              // global_addr[31:0]
        g0[3] = (uint32_t)((ga >> 32) & 0x01FFFFFFu)       // global_addr[56:32]
              | (2u << 30);                                // type = 2 ("image")

        i32x8 g1;
        g1[0] = (int)(2u << 16);                           // data_size = 4B
        g1[1] = (int)((uint32_t)NIN << 16);                // tensor_dim0[15:0] @ bits 63:48
        g1[2] = (int)(1u << 16);                           // tensor_dim0 hi=0 | tensor_dim1=1
        g1[3] = (int)((uint32_t)NIN << 16);                // tensor_dim1 hi=0 | tile_dim0=1470
        g1[4] = 1;                                         // tile_dim1 = 1, tile_dim2 = 0
        g1[5] = NIN;                                       // tensor_dim0_stride lo
        g1[6] = 0;                                         // stride hi / dim1_stride lo
        g1[7] = 0;

        i32x4 gz = {0, 0, 0, 0};
#if defined(__clang_major__) && (__clang_major__ >= 23)
        i32x8 gz8 = {0, 0, 0, 0, 0, 0, 0, 0};
        __builtin_amdgcn_tensor_load_to_lds(g0, g1, gz, gz, gz8, 0);
#else
        __builtin_amdgcn_tensor_load_to_lds(g0, g1, gz, gz, 0);
#endif
        __builtin_amdgcn_s_wait_tensorcnt(0);
    }
    __syncthreads();
    for (int i = t; i < NIN; i += 256)
        s_p[i] = 1.0f / (1.0f + expf(-s_p[i]));
#else
    for (int i = t; i < NIN; i += 256)
        s_p[i] = 1.0f / (1.0f + expf(-x[i]));
#endif
    __syncthreads();

    // ---------------- Phase 2: per-cell decode ----------------
    if (t < NCELLS) {
        const int h = t / GRID7, w = t % GRID7;
        const int base = t * NCH;

        // class argmax (first-max wins, matches jnp.argmax)
        float bestc = s_p[base];
        int   bcls  = 0;
        #pragma unroll
        for (int c = 1; c < NCLS; ++c) {
            const float v = s_p[base + c];
            if (v > bestc) { bestc = v; bcls = c; }
        }
        // best of 2 boxes (tie -> box 0)
        const float c0 = s_p[base + 20], c1 = s_p[base + 25];
        const int   b  = (c1 > c0) ? 1 : 0;
        const int   off = base + 20 + b * 5;
        const float conf = b ? c1 : c0;
        const float px = s_p[off + 1], py = s_p[off + 2];
        const float pw = s_p[off + 3], ph = s_p[off + 4];

        s_box[t * 6 + 0] = (px + (float)w) * STRIDE;
        s_box[t * 6 + 1] = (py + (float)h) * STRIDE;
        s_box[t * 6 + 2] = pw * (GRID7 * STRIDE);   // *448
        s_box[t * 6 + 3] = ph * (GRID7 * STRIDE);
        s_box[t * 6 + 4] = conf;
        s_box[t * 6 + 5] = (float)bcls;
        s_conf[t] = conf;                           // in (0,1)
    } else if (t < 64) {
        s_conf[t] = -1.0f;                          // pad: never beats a sigmoid
    }
    __syncthreads();

    // ---------------- Phase 3: stable-descending rank via WMMA ----------------
    // rank[j] = #{k : conf_k > conf_j} + #{k < j : conf_k == conf_j}
    // Column sums of a 0/1 matrix: D = ones(16x32) x G(32x16) on the matrix core.
    // Pad entries are -1 and real confs are in (0,1), so no k<NCELLS guard needed:
    // for k>=49, ck>cj and ck==cj are both false against any valid cj.
    if (t < 32) {                                   // wave 0, EXEC all ones
        const int lane  = t;
        const int n     = lane & 15;                // output column within tile
        const int khalf = (lane >> 4) << 4;         // 0 or 16: K sub-block this lane feeds

        v16h ones;
        #pragma unroll
        for (int i = 0; i < 16; ++i) ones[i] = (_Float16)1.0f;

        // Hoist this lane's 32 K-values into registers (reused by all 4 j-tiles).
        float ck[32];
        #pragma unroll
        for (int hh = 0; hh < 16; ++hh) {
            ck[hh]      = s_conf[khalf + hh];       // kc = 0 : k = khalf + hh
            ck[16 + hh] = s_conf[32 + khalf + hh];  // kc = 1 : k = 32 + khalf + hh
        }

        #pragma unroll
        for (int jt = 0; jt < 4; ++jt) {
            const int   j  = jt * 16 + n;
            const float cj = s_conf[j];
            v8f acc = {0.f, 0.f, 0.f, 0.f, 0.f, 0.f, 0.f, 0.f};
            #pragma unroll
            for (int kc = 0; kc < 2; ++kc) {
                v16h bm;
                #pragma unroll
                for (int hh = 0; hh < 16; ++hh) {
                    const int   k = kc * 32 + khalf + hh;
                    const float c = ck[kc * 16 + hh];
                    const bool  g = (c > cj) || ((c == cj) && (k < j));
                    bm[hh] = g ? (_Float16)1.0f : (_Float16)0.0f;
                }
                acc = __builtin_amdgcn_wmma_f32_16x16x32_f16(
                        false, ones, false, bm, (short)0, acc, false, false);
            }
            // every row of D equals the column sum; VGPR0, lanes 0-15 carry N=0..15
            if (lane < 16 && j < NCELLS)
                s_rank[j] = (int)(acc[0] + 0.5f);
        }
    }
    __syncthreads();

    // ---------------- Phase 4: scatter into sorted order ----------------
    if (t < NCELLS) {
        const int r = s_rank[t];                    // permutation of 0..48
        #pragma unroll
        for (int c = 0; c < 6; ++c)
            s_sb[r * 6 + c] = s_box[t * 6 + c];
    }
    __syncthreads();

    // ---------------- Phase 5: IoU matrix + keep init ----------------
    if (t < NCELLS)
        s_keep[t] = (s_sb[t * 6 + 4] > CONF_TH) ? 1 : 0;
    for (int p = t; p < NCELLS * NCELLS; p += 256) {
        const int i = p / NCELLS, j = p % NCELLS;
        const float xi = s_sb[i*6+0], yi = s_sb[i*6+1], wi = s_sb[i*6+2], hi = s_sb[i*6+3];
        const float xj = s_sb[j*6+0], yj = s_sb[j*6+1], wj = s_sb[j*6+2], hj = s_sb[j*6+3];
        const float ix0 = fmaxf(xi - wi * 0.5f, xj - wj * 0.5f);
        const float iy0 = fmaxf(yi - hi * 0.5f, yj - hj * 0.5f);
        const float ix1 = fminf(xi + wi * 0.5f, xj + wj * 0.5f);
        const float iy1 = fminf(yi + hi * 0.5f, yj + hj * 0.5f);
        const float iw  = fmaxf(ix1 - ix0, 0.0f);
        const float ih  = fmaxf(iy1 - iy0, 0.0f);
        const float inter = iw * ih;
        const float uni   = wi * hi + wj * hj - inter;
        s_iou[p] = inter / fmaxf(uni, 1e-9f);
    }
    __syncthreads();

    // ---------------- Phase 6: greedy NMS (sequential pivots, parallel columns) --
    for (int i = 0; i < NCELLS; ++i) {
        if (t < NCELLS && t > i) {
            if (s_keep[i] && (s_iou[i * NCELLS + t] > IOU_TH) &&
                (s_sb[i * 6 + 5] == s_sb[t * 6 + 5]))
                s_keep[t] = 0;
        }
        __syncthreads();                            // uniform: all 256 threads
    }

    // ---------------- Phase 7: masked output ----------------
    for (int i = t; i < NCELLS * 6; i += 256)
        out[i] = s_sb[i] * (s_keep[i / 6] ? 1.0f : 0.0f);
}

extern "C" void kernel_launch(void* const* d_in, const int* in_sizes, int n_in,
                              void* d_out, int out_size, void* d_ws, size_t ws_size,
                              hipStream_t stream)
{
    (void)in_sizes; (void)n_in; (void)out_size; (void)d_ws; (void)ws_size;
    const float* x   = (const float*)d_in[0];
    float*       out = (float*)d_out;
    hipLaunchKernelGGL(yolo_decode_nms_kernel, dim3(1), dim3(256), 0, stream, x, out);
}
